// LinearAttention_59502476919350
// MI455X (gfx1250) — matrix operand order repaired and checked by reading, not compiled, not run on hardware
//
#include <hip/hip_runtime.h>

typedef __attribute__((ext_vector_type(16))) __bf16 v16bf;
typedef __attribute__((ext_vector_type(8)))  __bf16 v8bf;
typedef __attribute__((ext_vector_type(8)))  float  v8f;
typedef __attribute__((ext_vector_type(4)))  unsigned int v4u;
typedef __attribute__((ext_vector_type(4)))  int v4i;
typedef __attribute__((ext_vector_type(8)))  int v8i;

#define B_   4
#define C_   256
#define N_   32768
#define E_   256
#define H_   8
#define HD_  32
#define GROUPS_ 32
#define CPG_ 8
#define MQKV_ 768

static __device__ __forceinline__ __bf16 f2bf(float f) {
  unsigned u = __builtin_bit_cast(unsigned, f);
  unsigned r = (u + 0x7FFFu + ((u >> 16) & 1u)) >> 16;   // round-to-nearest-even
  unsigned short s = (unsigned short)r;
  return __builtin_bit_cast(__bf16, s);
}

static __device__ __forceinline__ v16bf mk16(v8bf lo, v8bf hi) {
  v16bf a;
#pragma unroll
  for (int i = 0; i < 8; ++i) { a[i] = lo[i]; a[i + 8] = hi[i]; }
  return a;
}

// A-fragment of V_WMMA_F32_16X16X32_BF16: source row-major A[m][k], row stride ld.
static __device__ __forceinline__ v16bf frag_a(const __bf16* base, int ld, int kb, int lane) {
  int m = lane & 15, h = lane >> 4;
  const __bf16* p = base + m * ld + kb + 8 * h;
  v8bf lo = *reinterpret_cast<const v8bf*>(p);
  v8bf hi = *reinterpret_cast<const v8bf*>(p + 16);
  return mk16(lo, hi);
}

// B-fragment: source holds B transposed, bt[n][k], row stride ld.
static __device__ __forceinline__ v16bf frag_b(const __bf16* bt, int ld, int kb, int lane) {
  int n = lane & 15, h = lane >> 4;
  const __bf16* p = bt + n * ld + kb + 8 * h;
  v8bf lo = *reinterpret_cast<const v8bf*>(p);
  v8bf hi = *reinterpret_cast<const v8bf*>(p + 16);
  return mk16(lo, hi);
}

static __device__ __forceinline__ v8f wmma_bf16(v16bf a, v16bf b, v8f c) {
  return __builtin_amdgcn_wmma_f32_16x16x32_bf16(false, a, false, b, (short)0, c, false, false);
}

static __device__ __forceinline__ v8f vzero() {
  v8f z;
#pragma unroll
  for (int i = 0; i < 8; ++i) z[i] = 0.0f;
  return z;
}

// Issue a 2D TDM load: tile_d0 elements (bf16) x tile_d1 rows, row stride = stride
// elements, from global `src` into LDS byte offset `ldsaddr`.
static __device__ __forceinline__ void tdm_load_2d_bf16(const __bf16* src, unsigned ldsaddr,
                                                        unsigned tile_d0, unsigned tile_d1,
                                                        unsigned tensor_d0, unsigned tensor_d1,
                                                        unsigned stride) {
  unsigned long long ga = (unsigned long long)(size_t)src;
  v4u g0;
  g0[0] = 1u;                                           // count=1, user descriptor
  g0[1] = ldsaddr;                                      // lds_addr
  g0[2] = (unsigned)(ga & 0xFFFFFFFFu);                 // global_addr lo
  g0[3] = (unsigned)((ga >> 32) & 0x01FFFFFFu) | 0x80000000u;  // addr hi | type=2
  v8i g1;
  g1[0] = 0x00010000;                                   // data_size=1 (2 bytes)
  g1[1] = (int)(tensor_d0 << 16);                       // tensor_dim0[15:0]
  g1[2] = (int)(tensor_d1 << 16);                       // tensor_dim0 hi=0 | tensor_dim1 lo
  g1[3] = (int)(tile_d0 << 16);                         // tensor_dim1 hi=0 | tile_dim0
  g1[4] = (int)tile_d1;                                 // tile_dim1 | tile_dim2=0
  g1[5] = (int)stride;                                  // tensor_dim0_stride lo32
  g1[6] = 0;
  g1[7] = 0;
  v4i g2, g3;
#pragma unroll
  for (int i = 0; i < 4; ++i) { g2[i] = 0; g3[i] = 0; }
#if defined(__clang_major__) && (__clang_major__ >= 23)
  v8i g4;
#pragma unroll
  for (int i = 0; i < 8; ++i) g4[i] = 0;
  __builtin_amdgcn_tensor_load_to_lds(g0, g1, g2, g3, g4, 0);
#else
  __builtin_amdgcn_tensor_load_to_lds(g0, g1, g2, g3, 0);
#endif
}

// ---------------- Kernel 1: GroupNorm statistics ----------------
__global__ __launch_bounds__(256) void k_gn_stats(const float* __restrict__ q,
                                                  float* __restrict__ stats) {
  int bg = blockIdx.x;                 // 0 .. B*GROUPS-1
  int b = bg / GROUPS_, g = bg % GROUPS_;
  const float* base = q + ((size_t)b * C_ + g * CPG_) * N_;
  float s1 = 0.f, s2 = 0.f;
  for (int c = 0; c < CPG_; ++c) {
    const float* row = base + (size_t)c * N_;
    for (int n = threadIdx.x; n < N_; n += 256) {
      float v = row[n];
      s1 += v; s2 += v * v;
    }
  }
  __shared__ float r1[256], r2[256];
  r1[threadIdx.x] = s1; r2[threadIdx.x] = s2;
  __syncthreads();
  for (int s = 128; s > 0; s >>= 1) {
    if (threadIdx.x < s) { r1[threadIdx.x] += r1[threadIdx.x + s];
                           r2[threadIdx.x] += r2[threadIdx.x + s]; }
    __syncthreads();
  }
  if (threadIdx.x == 0) {
    float cnt  = (float)(CPG_ * N_);
    float mean = r1[0] / cnt;
    float var  = r2[0] / cnt - mean * mean;
    stats[bg * 2]     = mean;
    stats[bg * 2 + 1] = rsqrtf(var + 1e-6f);
  }
}

// ------- Kernel 2: zero dots, convert Wout and Wq|Wkv to bf16 -------
__global__ __launch_bounds__(256) void k_prep(const float* __restrict__ Wout,
                                              const float* __restrict__ Wq,
                                              const float* __restrict__ Wkv,
                                              float* __restrict__ dots,
                                              __bf16* __restrict__ woutb,
                                              __bf16* __restrict__ wqkvb) {
  int i = blockIdx.x * 256 + threadIdx.x;          // grid: 768*256 = 196608
  if (i < B_ * H_ * HD_ * HD_) dots[i] = 0.0f;
  if (i < C_ * E_) woutb[i] = f2bf(Wout[i]);
  if (i < MQKV_ * C_) {
    float w = (i < E_ * C_) ? Wq[i] : Wkv[i - E_ * C_];
    wqkvb[i] = f2bf(w);
  }
}

// ---------------- Kernel 3: fused groupnorm-apply + QKV projection ----------------
// A (128x256 bf16 weight slab) comes in via one TDM tensor_load_to_lds per block.
__global__ __launch_bounds__(256) void k_qkv(
    const float* __restrict__ q, const float* __restrict__ gamma, const float* __restrict__ beta,
    const __bf16* __restrict__ wqkvb, const float* __restrict__ bq,
    const float* __restrict__ bkv, const float* __restrict__ stats,
    __bf16* __restrict__ qkv) {
  int b  = blockIdx.z;
  int m0 = blockIdx.y * 128;
  int n0 = blockIdx.x * 64;
  int tid = threadIdx.x;
  int lane = tid & 31, wave = tid >> 5;
  int wm = wave & 3, wn = wave >> 2;

  __shared__ __align__(16) __bf16 aT[128 * 256];  // weight slab [m][k], 64 KB
  __shared__ __align__(16) __bf16 hT[64 * 256];   // h tile transposed [n][k], 32 KB
  __shared__ float sm[C_], sr[C_], sg[C_], sb[C_];

  // Kick off the TDM copy of the weight slab (wave 0 issues; EXEC ignored by TDM).
  if (tid < 32) {
    tdm_load_2d_bf16(wqkvb + (size_t)m0 * C_, (unsigned)(size_t)(&aT[0]),
                     /*tile_d0=*/C_, /*tile_d1=*/128,
                     /*tensor_d0=*/C_, /*tensor_d1=*/MQKV_, /*stride=*/C_);
  }
  {
    int c = tid;                       // 256 threads == C_
    int bg = b * GROUPS_ + c / CPG_;
    sm[c] = stats[bg * 2]; sr[c] = stats[bg * 2 + 1];
    sg[c] = gamma[c];      sb[c] = beta[c];
  }
  __syncthreads();
  // Normalized h tile, transposed: hT[n][k] (overlaps with the TDM transfer)
  for (int t = 0; t < 64; ++t) {
    int idx = t * 256 + tid;           // 0..16383
    int k = idx >> 6, n = idx & 63;
    float v = q[((size_t)b * C_ + k) * N_ + n0 + n];
    v = (v - sm[k]) * sr[k] * sg[k] + sb[k];
    hT[n * 256 + k] = f2bf(v);
  }
  if (tid < 32) __builtin_amdgcn_s_wait_tensorcnt(0);
  __syncthreads();

  v8f acc[2][2];
#pragma unroll
  for (int i = 0; i < 2; ++i)
#pragma unroll
    for (int j = 0; j < 2; ++j) acc[i][j] = vzero();

#pragma unroll
  for (int kb = 0; kb < C_; kb += 32) {
#pragma unroll
    for (int at = 0; at < 2; ++at) {
      v16bf af = frag_a(aT + (wm * 32 + at * 16) * 256, 256, kb, lane);
#pragma unroll
      for (int bt = 0; bt < 2; ++bt) {
        v16bf bf = frag_b(hT + (wn * 32 + bt * 16) * 256, 256, kb, lane);
        acc[at][bt] = wmma_bf16(af, bf, acc[at][bt]);
      }
    }
  }

  const float* bp = (m0 < E_) ? bq : (bkv - E_);   // block-uniform bias base
  int half = lane >> 4, nl = lane & 15;
#pragma unroll
  for (int at = 0; at < 2; ++at)
#pragma unroll
    for (int bt = 0; bt < 2; ++bt)
#pragma unroll
      for (int r = 0; r < 8; ++r) {
        int mg = m0 + wm * 32 + at * 16 + r + 8 * half;
        int ng = n0 + wn * 32 + bt * 16 + nl;
        qkv[((size_t)b * MQKV_ + mg) * N_ + ng] = f2bf(acc[at][bt][r] + bp[mg]);
      }
}

// ---------------- Kernel 4: dots = scale * qh @ kh^T (contract over N) ----------------
__global__ __launch_bounds__(256) void k_dots(const __bf16* __restrict__ qkv,
                                              float* __restrict__ dots) {
  int b = blockIdx.z, h = blockIdx.y;
  int wave = threadIdx.x >> 5, lane = threadIdx.x & 31;
  int nb0 = blockIdx.x * 4096 + wave * 512;
  const __bf16* qp = qkv + ((size_t)b * MQKV_ + h * HD_) * N_;
  const __bf16* kp = qkv + ((size_t)b * MQKV_ + E_ + h * HD_) * N_;

  v8f acc[2][2];
#pragma unroll
  for (int i = 0; i < 2; ++i)
#pragma unroll
    for (int j = 0; j < 2; ++j) acc[i][j] = vzero();

  for (int s = 0; s < 16; ++s) {
    int nb = nb0 + s * 32;
#pragma unroll
    for (int it = 0; it < 2; ++it) {
      v16bf af = frag_a(qp + (size_t)(it * 16) * N_, N_, nb, lane);
#pragma unroll
      for (int jt = 0; jt < 2; ++jt) {
        v16bf bf = frag_b(kp + (size_t)(jt * 16) * N_, N_, nb, lane);
        acc[it][jt] = wmma_bf16(af, bf, acc[it][jt]);
      }
    }
  }
  const float scale = 0.17677669529663687f;   // 32^-0.5
  int half = lane >> 4, nl = lane & 15;
  float* dbase = dots + (size_t)(b * H_ + h) * HD_ * HD_;
#pragma unroll
  for (int it = 0; it < 2; ++it)
#pragma unroll
    for (int jt = 0; jt < 2; ++jt)
#pragma unroll
      for (int r = 0; r < 8; ++r) {
        int i = it * 16 + r + 8 * half, j = jt * 16 + nl;
        atomicAdd(dbase + i * HD_ + j, acc[it][jt][r] * scale);
      }
}

// ---------------- Kernel 5: softmax over j (32) ----------------
__global__ void k_softmax(const float* __restrict__ dots, __bf16* __restrict__ attn) {
  int bh = blockIdx.x;                 // 0..31
  int i  = threadIdx.x;                // 32 lanes, one row each
  const float* row = dots + (size_t)bh * HD_ * HD_ + i * HD_;
  float mx = -3.4e38f;
#pragma unroll
  for (int j = 0; j < HD_; ++j) mx = fmaxf(mx, row[j]);
  float e[HD_]; float s = 0.f;
#pragma unroll
  for (int j = 0; j < HD_; ++j) { e[j] = __expf(row[j] - mx); s += e[j]; }
  float inv = 1.0f / s;
  __bf16* arow = attn + (size_t)bh * HD_ * HD_ + i * HD_;
#pragma unroll
  for (int j = 0; j < HD_; ++j) arow[j] = f2bf(e[j] * inv);
}

// ---------------- Kernel 6: (attn @ v) then Wout GEMM + bias + residual ----------------
__global__ __launch_bounds__(256) void k_out(
    const __bf16* __restrict__ qkv, const __bf16* __restrict__ attn,
    const __bf16* __restrict__ woutb, const float* __restrict__ bout,
    const float* __restrict__ qin, float* __restrict__ out) {
  int b  = blockIdx.y;
  int n0 = blockIdx.x * 64;
  int tid = threadIdx.x, lane = tid & 31, wave = tid >> 5;

  __shared__ __align__(16) __bf16 vT[64 * 256];    // v tile transposed [n][e]
  __shared__ __align__(16) __bf16 aoT[64 * 256];   // attn@v tile transposed [n][e]

  const __bf16* vbase = qkv + ((size_t)b * MQKV_ + 2 * E_) * N_;
  for (int t = 0; t < 64; ++t) {
    int idx = t * 256 + tid;           // 0..16383
    int e = idx >> 6, n = idx & 63;
    vT[n * 256 + e] = vbase[(size_t)e * N_ + n0 + n];
  }
  __syncthreads();

  // Step 1: one wave per head: ao = attn(32x32) @ v(32x64), single K=32 step
  {
    int h = wave;
    const __bf16* abase = attn + (size_t)(b * H_ + h) * HD_ * HD_;
    v8f acc[2][4];
#pragma unroll
    for (int it = 0; it < 2; ++it)
#pragma unroll
      for (int nt = 0; nt < 4; ++nt) acc[it][nt] = vzero();
#pragma unroll
    for (int it = 0; it < 2; ++it) {
      v16bf af = frag_a(abase + it * 16 * HD_, HD_, 0, lane);
#pragma unroll
      for (int nt = 0; nt < 4; ++nt) {
        v16bf bf = frag_b(vT + (nt * 16) * 256, 256, h * 32, lane);
        acc[it][nt] = wmma_bf16(af, bf, acc[it][nt]);
      }
    }
    int half = lane >> 4, nl = lane & 15;
#pragma unroll
    for (int it = 0; it < 2; ++it)
#pragma unroll
      for (int nt = 0; nt < 4; ++nt)
#pragma unroll
        for (int r = 0; r < 8; ++r) {
          int e = h * 32 + it * 16 + r + 8 * half;
          int n = nt * 16 + nl;
          aoT[n * 256 + e] = f2bf(acc[it][nt][r]);
        }
  }
  __syncthreads();

  // Step 2: out = Wout(256x256) @ ao(256x64) + bout + residual
  {
    v8f acc[2][4];
#pragma unroll
    for (int at = 0; at < 2; ++at)
#pragma unroll
      for (int nt = 0; nt < 4; ++nt) acc[at][nt] = vzero();

    for (int kb = 0; kb < E_; kb += 32) {
#pragma unroll
      for (int at = 0; at < 2; ++at) {
        v16bf af = frag_a(woutb + (size_t)(wave * 32 + at * 16) * 256, 256, kb, lane);
#pragma unroll
        for (int nt = 0; nt < 4; ++nt) {
          v16bf bf = frag_b(aoT + (nt * 16) * 256, 256, kb, lane);
          acc[at][nt] = wmma_bf16(af, bf, acc[at][nt]);
        }
      }
    }
    int half = lane >> 4, nl = lane & 15;
#pragma unroll
    for (int at = 0; at < 2; ++at)
#pragma unroll
      for (int nt = 0; nt < 4; ++nt)
#pragma unroll
        for (int r = 0; r < 8; ++r) {
          int c = wave * 32 + at * 16 + r + 8 * half;
          int n = n0 + nt * 16 + nl;
          size_t off = ((size_t)b * C_ + c) * N_ + n;
          out[off] = acc[at][nt][r] + bout[c] + qin[off];
        }
  }
}

extern "C" void kernel_launch(void* const* d_in, const int* in_sizes, int n_in,
                              void* d_out, int out_size, void* d_ws, size_t ws_size,
                              hipStream_t stream) {
  (void)in_sizes; (void)n_in; (void)out_size; (void)ws_size;
  const float* q     = (const float*)d_in[0];
  const float* gamma = (const float*)d_in[1];
  const float* beta  = (const float*)d_in[2];
  const float* Wq    = (const float*)d_in[3];
  const float* bq    = (const float*)d_in[4];
  const float* Wkv   = (const float*)d_in[5];
  const float* bkv   = (const float*)d_in[6];
  const float* Wout  = (const float*)d_in[7];
  const float* bout  = (const float*)d_in[8];
  float* out = (float*)d_out;

  char* ws = (char*)d_ws;
  float*  stats = (float*)(ws);                    //   4 KB
  float*  dots  = (float*)(ws + 4096);             // 128 KB
  __bf16* attn  = (__bf16*)(ws + 135168);          //  64 KB
  __bf16* woutb = (__bf16*)(ws + 200704);          // 128 KB
  __bf16* wqkvb = (__bf16*)(ws + 331776);          // 384 KB (Wq|Wkv bf16)
  __bf16* qkvb  = (__bf16*)(ws + (1u << 20));      // 192 MB (bf16 qp|k|v)

  k_gn_stats<<<dim3(B_ * GROUPS_), 256, 0, stream>>>(q, stats);
  k_prep    <<<dim3(768),          256, 0, stream>>>(Wout, Wq, Wkv, dots, woutb, wqkvb);
  k_qkv     <<<dim3(N_ / 64, MQKV_ / 128, B_), 256, 0, stream>>>(
                q, gamma, beta, wqkvb, bq, bkv, stats, qkvb);
  k_dots    <<<dim3(8, H_, B_),    256, 0, stream>>>(qkvb, dots);
  k_softmax <<<dim3(B_ * H_),       32, 0, stream>>>(dots, attn);
  k_out     <<<dim3(N_ / 64, B_),  256, 0, stream>>>(qkvb, attn, woutb, bout, q, out);
}
